// FusionFeedForward_40973988004146
// MI455X (gfx1250) — matrix operand order repaired and checked by reading, not compiled, not run on hardware
//
#include <hip/hip_runtime.h>
#include <stdint.h>

typedef __bf16 bf16_t;
typedef __attribute__((ext_vector_type(16))) __bf16 v16bf;
typedef __attribute__((ext_vector_type(8)))  __bf16 v8bf;
typedef __attribute__((ext_vector_type(8)))  float  v8f;
typedef int v4i_vs __attribute__((vector_size(16)));   // matches builtin param pointee

#if defined(__has_builtin)
#if __has_builtin(__builtin_amdgcn_global_load_async_to_lds_b128)
#define HAVE_ASYNC_LDS 1
#endif
#if __has_builtin(__builtin_amdgcn_s_wait_asynccnt)
#define HAVE_WAIT_ASYNC_BUILTIN 1
#endif
#endif

__device__ __forceinline__ void wait_async0() {
#if defined(HAVE_ASYNC_LDS)
#if defined(HAVE_WAIT_ASYNC_BUILTIN)
  __builtin_amdgcn_s_wait_asynccnt(0);
#else
  asm volatile("s_wait_asynccnt 0x0" ::: "memory");
#endif
#endif
}

// Stage one 128x32 bf16 tile (8KB) into LDS: 256 threads x 32B each.
__device__ __forceinline__ void stage_tile(const bf16_t* __restrict__ g, int ld,
                                           size_t base_row, int k0,
                                           bf16_t* __restrict__ lds, int tid) {
  int r = tid >> 1, hs = (tid & 1) * 16;
  const bf16_t* src = g + (base_row + r) * (size_t)ld + k0 + hs;
  bf16_t* dst = lds + r * 32 + hs;
#if defined(HAVE_ASYNC_LDS)
  auto gp = (__attribute__((address_space(1))) v4i_vs*)(uintptr_t)src;
  auto lp = (__attribute__((address_space(3))) v4i_vs*)(uintptr_t)dst;
  __builtin_amdgcn_global_load_async_to_lds_b128(gp, lp, 0, 0);
  __builtin_amdgcn_global_load_async_to_lds_b128(gp, lp, 16, 0);
#else
  const float4* s = (const float4*)src;
  float4* d = (float4*)dst;
  d[0] = s[0]; d[1] = s[1];
#endif
}

// ---------------- conversions / staging ----------------

__global__ __launch_bounds__(256) void cvt_x_kernel(const float* __restrict__ x,
                                                    bf16_t* __restrict__ xb, int n) {
  int i = blockIdx.x * blockDim.x + threadIdx.x;
  if (i < n) xb[i] = (bf16_t)x[i];
}

// W1 (1960,512) -> W1b (2048,512) zero-padded rows
__global__ __launch_bounds__(256) void cvt_w1_kernel(const float* __restrict__ W1,
                                                     bf16_t* __restrict__ W1b) {
  int i = blockIdx.x * blockDim.x + threadIdx.x;
  if (i >= 2048 * 512) return;
  int row = i / 512, k = i % 512;
  W1b[i] = (row < 1960) ? (bf16_t)W1[row * 512 + k] : (bf16_t)0.0f;
}

// W2 (512,1960) -> W2b (512,1984) zero-padded cols
__global__ __launch_bounds__(256) void cvt_w2_kernel(const float* __restrict__ W2,
                                                     bf16_t* __restrict__ W2b) {
  int i = blockIdx.x * blockDim.x + threadIdx.x;
  if (i >= 512 * 1984) return;
  int row = i / 1984, col = i % 1984;
  W2b[i] = (col < 1960) ? (bf16_t)W2[(size_t)row * 1960 + col] : (bf16_t)0.0f;
}

// ---------------- bf16 WMMA GEMM: C[M,N] = A[M,K] * B[N,K]^T + bias ----------------
// Tile 128(M) x 128(N), K-step 32, 8 wave32 waves; each wave a 32x64 subtile
// (2x4 accumulators of 16x16). Double-buffered LDS filled with async-to-LDS.

__global__ __launch_bounds__(256)
void gemm_bf16_wmma(const bf16_t* __restrict__ A, int lda,
                    const bf16_t* __restrict__ B, int ldb,
                    const float* __restrict__ bias, int bias_len,
                    float* __restrict__ C, int ldc, int K) {
  __shared__ bf16_t As[2][128 * 32];
  __shared__ bf16_t Bs[2][128 * 32];

  const int tid   = threadIdx.x;
  const int lane  = tid & 31;
  const int wave  = tid >> 5;     // 0..7
  const int wm    = wave & 3;     // row group of 32
  const int wn    = wave >> 2;    // col group of 64
  const int lrow  = lane & 15;
  const int lhalf = lane >> 4;    // 0/1
  const size_t block_m = (size_t)blockIdx.x * 128;
  const size_t block_n = (size_t)blockIdx.y * 128;

  v8f acc[2][4] = {};

  const int nk = K / 32;
  stage_tile(A, lda, block_m, 0, As[0], tid);
  stage_tile(B, ldb, block_n, 0, Bs[0], tid);

  for (int ki = 0; ki < nk; ++ki) {
    const int cur = ki & 1;
    wait_async0();       // this wave's staged tile has landed in LDS
    __syncthreads();     // all waves' tiles visible
    if (ki + 1 < nk) {   // prefetch next K-step into the other buffer
      stage_tile(A, lda, block_m, (ki + 1) * 32, As[cur ^ 1], tid);
      stage_tile(B, ldb, block_n, (ki + 1) * 32, Bs[cur ^ 1], tid);
    }

    v16bf afrag[2], bfrag[4];
#pragma unroll
    for (int rb = 0; rb < 2; ++rb) {
      const bf16_t* ap = As[cur] + (wm * 32 + rb * 16 + lrow) * 32;
      v8bf lo = *(const v8bf*)(ap + lhalf * 8);
      v8bf hi = *(const v8bf*)(ap + 16 + lhalf * 8);
#pragma unroll
      for (int i = 0; i < 8; ++i) { afrag[rb][i] = lo[i]; afrag[rb][i + 8] = hi[i]; }
    }
#pragma unroll
    for (int cb = 0; cb < 4; ++cb) {
      const bf16_t* bp = Bs[cur] + (wn * 64 + cb * 16 + lrow) * 32;
      v8bf lo = *(const v8bf*)(bp + lhalf * 8);
      v8bf hi = *(const v8bf*)(bp + 16 + lhalf * 8);
#pragma unroll
      for (int i = 0; i < 8; ++i) { bfrag[cb][i] = lo[i]; bfrag[cb][i + 8] = hi[i]; }
    }
#pragma unroll
    for (int rb = 0; rb < 2; ++rb)
#pragma unroll
      for (int cb = 0; cb < 4; ++cb)
        acc[rb][cb] = __builtin_amdgcn_wmma_f32_16x16x32_bf16(
            false, afrag[rb], false, bfrag[cb], (short)0, acc[rb][cb], false, false);
  }

#pragma unroll
  for (int rb = 0; rb < 2; ++rb)
#pragma unroll
    for (int cb = 0; cb < 4; ++cb) {
      size_t col = block_n + wn * 64 + cb * 16 + lrow;
      float bv = ((int)col < bias_len) ? bias[col] : 0.0f;
      size_t mbase = block_m + wm * 32 + rb * 16 + lhalf * 8;
#pragma unroll
      for (int v = 0; v < 8; ++v)
        C[(mbase + v) * (size_t)ldc + col] = acc[rb][cb][v] + bv;
    }
}

// ---------------- fold (gather, normalized): h(row,2048) -> fn (16,40,60,108) ----------------

__global__ __launch_bounds__(256) void fold_norm_kernel(const float* __restrict__ h,
                                                        float* __restrict__ fn) {
  int idx = blockIdx.x * blockDim.x + threadIdx.x;
  if (idx >= 16 * 40 * 60 * 108) return;
  int px_ = idx % 108; int t = idx / 108;
  int py_ = t % 60; t /= 60;
  int c = t % 40; int bp = t / 40;
  int py = py_ + 3, px = px_ + 3;

  int cy = 0, cx = 0;
  for (int ky = py % 3; ky < 7; ky += 3) { int oy = (py - ky) / 3; if (oy >= 0 && oy < 20) cy++; }
  for (int kx = px % 3; kx < 7; kx += 3) { int ox = (px - kx) / 3; if (ox >= 0 && ox < 36) cx++; }

  float acc = 0.0f;
  for (int ky = py % 3; ky < 7; ky += 3) {
    int oy = (py - ky) / 3;
    if (oy < 0 || oy >= 20) continue;
    for (int kx = px % 3; kx < 7; kx += 3) {
      int ox = (px - kx) / 3;
      if (ox < 0 || ox >= 36) continue;
      int l = oy * 36 + ox;
      int hid = c * 49 + ky * 7 + kx;
      size_t row = (size_t)bp * 720 + l;   // == b*5760 + n
      acc += h[row * 2048 + hid];
    }
  }
  fn[idx] = acc / (float)(cy * cx);
}

// ---------------- unfold + exact GELU + bf16 pack (K padded to 1984) ----------------

__global__ __launch_bounds__(256) void unfold_gelu_kernel(const float* __restrict__ fn,
                                                          bf16_t* __restrict__ yb) {
  size_t idx = (size_t)blockIdx.x * blockDim.x + threadIdx.x;
  if (idx >= (size_t)11520 * 1984) return;
  int col = (int)(idx % 1984);
  int row = (int)(idx / 1984);
  bf16_t out = (bf16_t)0.0f;
  if (col < 1960) {
    int bp = row / 720, l = row % 720;
    int oy = l / 36, ox = l % 36;
    int c = col / 49, r = col % 49;
    int ky = r / 7, kx = r % 7;
    int py = ky + 3 * oy, px = kx + 3 * ox;
    float v = 0.0f;
    if (py >= 3 && py < 63 && px >= 3 && px < 111)
      v = fn[((size_t)(bp * 40 + c) * 60 + (py - 3)) * 108 + (px - 3)];
    float g = 0.5f * v * (1.0f + erff(v * 0.70710678118654752f));  // exact GELU
    out = (bf16_t)g;
  }
  yb[idx] = out;
}

// ---------------- launch ----------------

extern "C" void kernel_launch(void* const* d_in, const int* in_sizes, int n_in,
                              void* d_out, int out_size, void* d_ws, size_t ws_size,
                              hipStream_t stream) {
  const float* x  = (const float*)d_in[0];   // (2,5760,512)
  const float* W1 = (const float*)d_in[1];   // (1960,512)
  const float* b1 = (const float*)d_in[2];   // (1960)
  const float* W2 = (const float*)d_in[3];   // (512,1960)
  const float* b2 = (const float*)d_in[4];   // (512)
  float* out = (float*)d_out;                // (2,5760,512)

  char* ws = (char*)d_ws;
  size_t off = 0;
  auto carve = [&](size_t bytes) { size_t o = off; off += (bytes + 255) & ~(size_t)255; return o; };
  size_t xb_off  = carve((size_t)11520 * 512 * 2);
  size_t w1b_off = carve((size_t)2048 * 512 * 2);
  size_t h_off   = carve((size_t)11520 * 2048 * 4);   // reused later for yb (bf16, smaller)
  size_t fn_off  = carve((size_t)16 * 40 * 60 * 108 * 4);
  size_t w2b_off = carve((size_t)512 * 1984 * 2);

  bf16_t* xb  = (bf16_t*)(ws + xb_off);
  bf16_t* W1b = (bf16_t*)(ws + w1b_off);
  float*  h   = (float*)(ws + h_off);
  float*  fn  = (float*)(ws + fn_off);
  bf16_t* W2b = (bf16_t*)(ws + w2b_off);
  bf16_t* yb  = (bf16_t*)(ws + h_off);     // alias: h is dead once fn is built

  // 1) stage bf16 operands
  {
    int n = 11520 * 512;
    cvt_x_kernel<<<(n + 255) / 256, 256, 0, stream>>>(x, xb, n);
    cvt_w1_kernel<<<(2048 * 512 + 255) / 256, 256, 0, stream>>>(W1, W1b);
    cvt_w2_kernel<<<(512 * 1984 + 255) / 256, 256, 0, stream>>>(W2, W2b);
  }
  // 2) GEMM1: h(11520,2048) = xb(11520,512) * W1b^T + b1  (padded cols -> exactly 0)
  gemm_bf16_wmma<<<dim3(11520 / 128, 2048 / 128), 256, 0, stream>>>(
      xb, 512, W1b, 512, b1, 1960, h, 2048, 512);
  // 3) normalized fold (gather, no atomics)
  fold_norm_kernel<<<(16 * 40 * 60 * 108 + 255) / 256, 256, 0, stream>>>(h, fn);
  // 4) unfold + GELU -> bf16 A operand for GEMM2 (K padded to 1984)
  {
    size_t n = (size_t)11520 * 1984;
    unfold_gelu_kernel<<<(unsigned)((n + 255) / 256), 256, 0, stream>>>(fn, yb);
  }
  // 5) GEMM2: out(11520,512) = yb(11520,1984) * W2b^T + b2
  gemm_bf16_wmma<<<dim3(11520 / 128, 512 / 128), 256, 0, stream>>>(
      yb, 1984, W2b, 1984, b2, 512, out, 512, 1984);
}
